// GraphSAGEEncoder_18957985644790
// MI455X (gfx1250) — matrix operand order repaired and checked by reading, not compiled, not run on hardware
//
#include <hip/hip_runtime.h>
#include <hip/hip_fp16.h>

typedef __attribute__((ext_vector_type(16))) _Float16 v16h;
typedef __attribute__((ext_vector_type(8)))  _Float16 v8h;
typedef __attribute__((ext_vector_type(8)))  float    v8f;

#define DF 64

// ---------------- prep kernels ----------------

__global__ void zero_f32(float* __restrict__ p, long long n) {
  long long i = (long long)blockIdx.x * blockDim.x + threadIdx.x;
  long long stride = (long long)gridDim.x * blockDim.x;
  for (; i < n; i += stride) p[i] = 0.0f;
}

__global__ void degree_kernel(const int* __restrict__ dst, float* __restrict__ deg, int E) {
  int e = blockIdx.x * blockDim.x + threadIdx.x;
  if (e < E) unsafeAtomicAdd(&deg[dst[e]], 1.0f);  // hardware global_atomic_add_f32
}

__global__ void invdeg_kernel(float* __restrict__ deg, int N) {
  int i = blockIdx.x * blockDim.x + threadIdx.x;
  if (i < N) deg[i] = 1.0f / fmaxf(deg[i], 1.0f);
}

__global__ void cvt_f32_to_f16(const float* __restrict__ in, _Float16* __restrict__ out, long long n) {
  long long i = (long long)blockIdx.x * blockDim.x + threadIdx.x;
  if (i < n) out[i] = (_Float16)in[i];
}

// scatter-mean part 1: agg[dst] += feat[src]; 16 lanes per edge, float4 per lane
__global__ void scatter_add_kernel(const float* __restrict__ feat,
                                   const int* __restrict__ src,
                                   const int* __restrict__ dst,
                                   float* __restrict__ agg, int E) {
  long long t = (long long)blockIdx.x * blockDim.x + threadIdx.x;
  int e = (int)(t >> 4);
  if (e >= E) return;
  int q = ((int)t & 15) * 4;
  int s = src[e], d = dst[e];
  const float4 v = *(const float4*)(feat + (long long)s * DF + q);
  float* o = agg + (long long)d * DF + q;
  unsafeAtomicAdd(o + 0, v.x);
  unsafeAtomicAdd(o + 1, v.y);
  unsafeAtomicAdd(o + 2, v.z);
  unsafeAtomicAdd(o + 3, v.w);
}

// scatter-mean part 2: scale by 1/deg and convert to f16 for WMMA A operand
__global__ void scale_cvt_kernel(const float* __restrict__ agg, const float* __restrict__ invdeg,
                                 _Float16* __restrict__ out, long long n) {
  long long t = (long long)blockIdx.x * blockDim.x + threadIdx.x;
  if (t >= n) return;
  int node = (int)(t >> 6);
  out[t] = (_Float16)(agg[t] * invdeg[node]);
}

// ---------------- fused SAGE GEMM: out = agg@Wl^T + b + x@Wr^T (opt relu) ----------------
// One wave computes a 16x64 output tile using v_wmma_f32_16x16x32_f16.
// A layout (16-bit, 16x32):  lane: row=lane&15, hi=lane>>4;  a[j]=A[row][k0+hi*8+j], a[8+j]=A[row][k0+16+hi*8+j]
// B layout (16-bit, 32x16):  lane: col=lane&15, hi=lane>>4;  b[j]=B[k0+hi*8+j][col], b[8+j]=B[k0+16+hi*8+j][col]  (B=W^T -> W[col][k])
// C/D layout (f32 16x16):    acc[r] at lane -> (row0 + r + hi*8, n0 + (lane&15))
__global__ __launch_bounds__(256)
void sage_gemm_wmma(const _Float16* __restrict__ Ah,   // mean-agg features, f16 [N,64]
                    const _Float16* __restrict__ Xh,   // root features,     f16 [N,64]
                    const _Float16* __restrict__ Wl,   // f16 [64 out][64 in]
                    const _Float16* __restrict__ Wr,   // f16 [64 out][64 in]
                    const float* __restrict__ bias,    // f32 [64]
                    float* __restrict__ out,           // f32 [N,64]
                    int N, int doRelu) {
  const int lane = threadIdx.x & 31;
  const int wave = threadIdx.x >> 5;
  const int numTiles = (N + 15) >> 4;
  const int tile = blockIdx.x * 8 + wave;      // wave-uniform -> full-EXEC WMMA
  if (tile >= numTiles) return;
  const int row0 = tile * 16;

  const int r  = lane & 15;
  const int hi = lane >> 4;
  const int rowA = min(row0 + r, N - 1);       // clamp for (unused here) ragged tail

  v8f acc[4] = {};

#pragma unroll
  for (int k0 = 0; k0 < DF; k0 += 32) {
    const _Float16* pa = Ah + (long long)rowA * DF + k0 + hi * 8;
    const _Float16* px = Xh + (long long)rowA * DF + k0 + hi * 8;
    v8h a0 = *(const v8h*)(pa);
    v8h a1 = *(const v8h*)(pa + 16);
    v8h x0 = *(const v8h*)(px);
    v8h x1 = *(const v8h*)(px + 16);
    v16h aA, aX;
#pragma unroll
    for (int j = 0; j < 8; ++j) { aA[j] = a0[j]; aA[8 + j] = a1[j]; aX[j] = x0[j]; aX[8 + j] = x1[j]; }

#pragma unroll
    for (int t = 0; t < 4; ++t) {
      const int n0 = t * 16;
      const _Float16* pl = Wl + (long long)(n0 + r) * DF + k0 + hi * 8;
      const _Float16* pr = Wr + (long long)(n0 + r) * DF + k0 + hi * 8;
      v8h bl0 = *(const v8h*)(pl);
      v8h bl1 = *(const v8h*)(pl + 16);
      v8h br0 = *(const v8h*)(pr);
      v8h br1 = *(const v8h*)(pr + 16);
      v16h bL, bR;
#pragma unroll
      for (int j = 0; j < 8; ++j) { bL[j] = bl0[j]; bL[8 + j] = bl1[j]; bR[j] = br0[j]; bR[8 + j] = br1[j]; }
      acc[t] = __builtin_amdgcn_wmma_f32_16x16x32_f16(false, aA, false, bL, (short)0, acc[t], false, false);
      acc[t] = __builtin_amdgcn_wmma_f32_16x16x32_f16(false, aX, false, bR, (short)0, acc[t], false, false);
    }
  }

  // bias + optional relu + store
#pragma unroll
  for (int t = 0; t < 4; ++t) {
    const int n0 = t * 16;
    const float bv = bias[n0 + r];
#pragma unroll
    for (int rr = 0; rr < 8; ++rr) {
      const int row = row0 + rr + hi * 8;
      if (row < N) {
        float v = acc[t][rr] + bv;
        if (doRelu) v = fmaxf(v, 0.0f);
        out[(long long)row * DF + (n0 + r)] = v;
      }
    }
  }
}

// ---------------- host orchestration ----------------

static inline size_t align256(size_t x) { return (x + 255) & ~(size_t)255; }

extern "C" void kernel_launch(void* const* d_in, const int* in_sizes, int n_in,
                              void* d_out, int out_size, void* d_ws, size_t ws_size,
                              hipStream_t stream) {
  const float* x   = (const float*)d_in[0];
  const int*   ei  = (const int*)d_in[1];
  const float* W1l = (const float*)d_in[2];
  const float* b1  = (const float*)d_in[3];
  const float* W1r = (const float*)d_in[4];
  const float* W2l = (const float*)d_in[5];
  const float* b2  = (const float*)d_in[6];
  const float* W2r = (const float*)d_in[7];
  float* out = (float*)d_out;

  const int N = in_sizes[0] / DF;
  const int E = in_sizes[1] / 2;
  const int* src = ei;
  const int* dst = ei + E;
  const long long nf = (long long)N * DF;

  // workspace carve-up
  char* ws = (char*)d_ws;
  size_t off = 0;
  float*    deg  = (float*)(ws + off);    off = align256(off + (size_t)N * 4);
  float*    agg  = (float*)(ws + off);    off = align256(off + (size_t)nf * 4);
  float*    h    = (float*)(ws + off);    off = align256(off + (size_t)nf * 4);
  _Float16* xh   = (_Float16*)(ws + off); off = align256(off + (size_t)nf * 2);
  _Float16* aggh = (_Float16*)(ws + off); off = align256(off + (size_t)nf * 2);
  _Float16* w16  = (_Float16*)(ws + off); off = align256(off + (size_t)4 * DF * DF * 2);
  _Float16* W1lh = w16;
  _Float16* W1rh = w16 + DF * DF;
  _Float16* W2lh = w16 + 2 * DF * DF;
  _Float16* W2rh = w16 + 3 * DF * DF;
  (void)ws_size; (void)n_in; (void)out_size;

  const int B = 256;
  const int zeroBlocks = 2048;
  const int nfBlocks   = (int)((nf + B - 1) / B);
  const int eBlocks    = (E + B - 1) / B;
  const int sBlocks    = (int)(((long long)E * 16 + B - 1) / B);
  const int nBlocks    = (N + B - 1) / B;
  const int numTiles   = (N + 15) >> 4;
  const int gemmBlocks = (numTiles + 7) / 8;

  // degrees -> inverse degrees
  zero_f32<<<zeroBlocks, B, 0, stream>>>(deg, N);
  degree_kernel<<<eBlocks, B, 0, stream>>>(dst, deg, E);
  invdeg_kernel<<<nBlocks, B, 0, stream>>>(deg, N);

  // f16 weights (once per call)
  cvt_f32_to_f16<<<(DF * DF + B - 1) / B, B, 0, stream>>>(W1l, W1lh, DF * DF);
  cvt_f32_to_f16<<<(DF * DF + B - 1) / B, B, 0, stream>>>(W1r, W1rh, DF * DF);
  cvt_f32_to_f16<<<(DF * DF + B - 1) / B, B, 0, stream>>>(W2l, W2lh, DF * DF);
  cvt_f32_to_f16<<<(DF * DF + B - 1) / B, B, 0, stream>>>(W2r, W2rh, DF * DF);

  // ---- layer 1 ----
  zero_f32<<<zeroBlocks, B, 0, stream>>>(agg, nf);
  cvt_f32_to_f16<<<nfBlocks, B, 0, stream>>>(x, xh, nf);
  scatter_add_kernel<<<sBlocks, B, 0, stream>>>(x, src, dst, agg, E);
  scale_cvt_kernel<<<nfBlocks, B, 0, stream>>>(agg, deg, aggh, nf);
  sage_gemm_wmma<<<gemmBlocks, B, 0, stream>>>(aggh, xh, W1lh, W1rh, b1, h, N, 1);

  // ---- layer 2 ----
  zero_f32<<<zeroBlocks, B, 0, stream>>>(agg, nf);
  cvt_f32_to_f16<<<nfBlocks, B, 0, stream>>>(h, xh, nf);
  scatter_add_kernel<<<sBlocks, B, 0, stream>>>(h, src, dst, agg, E);
  scale_cvt_kernel<<<nfBlocks, B, 0, stream>>>(agg, deg, aggh, nf);
  sage_gemm_wmma<<<gemmBlocks, B, 0, stream>>>(aggh, xh, W2lh, W2rh, b2, out, N, 0);
}
